// HierarchicalTreeLSTMs_44186623541725
// MI455X (gfx1250) — matrix operand order repaired and checked by reading, not compiled, not run on hardware
//
#include <hip/hip_runtime.h>

// ---------------- CDNA5 / gfx1250 types ----------------
typedef __bf16 bf16;
typedef __attribute__((ext_vector_type(16))) __bf16 v16bf;
typedef __attribute__((ext_vector_type(8)))  __bf16 v8bf;
typedef __attribute__((ext_vector_type(4)))  __bf16 v4bf;
typedef __attribute__((ext_vector_type(8)))  float  v8f;
typedef __attribute__((ext_vector_type(4)))  float  v4f;

#define TREES 16
#define NODES 21845   // (4^8 - 1) / 3

union Frag { v16bf v; v8bf h[2]; };

static __device__ __forceinline__ bf16 f2bf(float f) {
  unsigned u = __builtin_bit_cast(unsigned, f);
  unsigned r = u + 0x7FFFu + ((u >> 16) & 1u);   // round-to-nearest-even
  return __builtin_bit_cast(bf16, (unsigned short)(r >> 16));
}

static __device__ __forceinline__ v8f wmma_bf16(v16bf a, v16bf b, v8f c) {
  // D = A(16x32 bf16) * B(32x16 bf16) + C(16x16 f32)
  return __builtin_amdgcn_wmma_f32_16x16x32_bf16(false, a, false, b, (short)0, c,
                                                 false, false);
}

// A fragment (16x32 bf16) from a row-major LDS tile bf16[16][128].
// Lane m (0..15)/group g: elems e<8 -> K=kt*32+8g+e ; e>=8 -> K=kt*32+16+8g+(e-8)
static __device__ __forceinline__ v16bf load_a_lds(const bf16* t, int kt, int lane) {
  const int m = lane & 15, g = lane >> 4;
  const bf16* p = t + m * 128 + kt * 32 + 8 * g;
  Frag a;
  a.h[0] = *(const v8bf*)p;          // ds_load_b128
  a.h[1] = *(const v8bf*)(p + 16);   // ds_load_b128
  return a.v;
}

// B fragment (32x16 bf16) from row-major global weights W[row][kstride].
// B(k,n) = W[row0+n][k]; lane n/group g: elems e<8 -> K=kt*32+8g+e ; e>=8 -> K=kt*32+16+8g+(e-8).
static __device__ __forceinline__ v16bf load_b_w(const bf16* __restrict__ w, int row0,
                                                 int kt, int kstride, int lane) {
  const int n = lane & 15, g = lane >> 4;
  const bf16* p = w + (long)(row0 + n) * kstride + kt * 32 + 8 * g;
  Frag b;
  b.h[0] = *(const v8bf*)p;
  b.h[1] = *(const v8bf*)(p + 16);
  return b.v;
}

static __device__ __forceinline__ float sigm(float x)  { return 1.0f / (1.0f + __expf(-x)); }
static __device__ __forceinline__ float tanhf_(float x){ return 2.0f * sigm(2.0f * x) - 1.0f; }

// Stage a 16x128 f32 tile (head or child-c rows of this node tile) into LDS as bf16.
static __device__ __forceinline__ void stage_tile(bf16* dst, const float* __restrict__ src,
                                                  int rbase, int lshift, int nodeOff,
                                                  int isChild, int childIdx, int lane) {
  const int nmask = (1 << lshift) - 1;
#pragma unroll
  for (int v = 0; v < 16; ++v) {
    const int r = rbase + v;
    const int t = r >> lshift;          // tree
    const int i = r & nmask;            // node-in-level
    const int node = isChild ? (nodeOff + 4 * i + childIdx) : (nodeOff + i);
    const float* p = src + ((long)t * NODES + node) * 128 + lane * 4;
    const v4f f = *(const v4f*)p;       // global_load_b128, coalesced per row
    v4bf q = { f2bf(f.x), f2bf(f.y), f2bf(f.z), f2bf(f.w) };
    *(v4bf*)(dst + v * 128 + lane * 4) = q;   // ds_store_b64
  }
}

// One LSTM step for a 16-row tile. sIn: x_t (16x128 bf16 LDS), sHin: h_{t-1} (or null),
// sHout: receives h_t. c state in registers (cst[8] = 8x v8f). bsum = bih+bhh (fused).
static __device__ __forceinline__ void lstm_step(const bf16* sIn, const bf16* sHin, bf16* sHout,
                                                 const bf16* __restrict__ Wih,
                                                 const bf16* __restrict__ Whh,
                                                 const float* __restrict__ bsum,
                                                 v8f* cst, int lane) {
  const int n = lane & 15, g = lane >> 4;
#pragma unroll
  for (int jt = 0; jt < 8; ++jt) {
    const int col = jt * 16 + n;
    const float b0 = bsum[0 * 128 + col];
    const float b1 = bsum[1 * 128 + col];
    const float b2 = bsum[2 * 128 + col];
    const float b3 = bsum[3 * 128 + col];
    v8f a0 = {}, a1 = {}, a2 = {}, a3 = {};   // zero C -> inline SRC2=0 on first WMMA
#pragma unroll
    for (int kt = 0; kt < 4; ++kt) {
      const v16bf a = load_a_lds(sIn, kt, lane);
      a0 = wmma_bf16(a, load_b_w(Wih, 0 * 128 + jt * 16, kt, 128, lane), a0);
      a1 = wmma_bf16(a, load_b_w(Wih, 1 * 128 + jt * 16, kt, 128, lane), a1);
      a2 = wmma_bf16(a, load_b_w(Wih, 2 * 128 + jt * 16, kt, 128, lane), a2);
      a3 = wmma_bf16(a, load_b_w(Wih, 3 * 128 + jt * 16, kt, 128, lane), a3);
    }
    if (sHin) {
#pragma unroll
      for (int kt = 0; kt < 4; ++kt) {
        const v16bf a = load_a_lds(sHin, kt, lane);
        a0 = wmma_bf16(a, load_b_w(Whh, 0 * 128 + jt * 16, kt, 128, lane), a0);
        a1 = wmma_bf16(a, load_b_w(Whh, 1 * 128 + jt * 16, kt, 128, lane), a1);
        a2 = wmma_bf16(a, load_b_w(Whh, 2 * 128 + jt * 16, kt, 128, lane), a2);
        a3 = wmma_bf16(a, load_b_w(Whh, 3 * 128 + jt * 16, kt, 128, lane), a3);
      }
    }
    v8f c = cst[jt];
#pragma unroll
    for (int v = 0; v < 8; ++v) {
      const float iv = sigm(a0[v] + b0);
      const float fv = sigm(a1[v] + b1);
      const float gv = tanhf_(a2[v] + b2);
      const float ov = sigm(a3[v] + b3);
      const float cv = fv * c[v] + iv * gv;
      c[v] = cv;
      // C-layout element (m = v + 8g, n) -> transpose to row-major h tile via LDS
      sHout[(v + 8 * g) * 128 + jt * 16 + n] = f2bf(ov * tanhf_(cv));
    }
    cst[jt] = c;
  }
}

__global__ __launch_bounds__(256) void convert_w_kernel(const float* __restrict__ src,
                                                        bf16* __restrict__ dst, int n) {
  const int i = blockIdx.x * 256 + threadIdx.x;
  if (i < n) dst[i] = f2bf(src[i]);
}

__global__ __launch_bounds__(256) void fuse_bias_kernel(const float* __restrict__ a,
                                                        const float* __restrict__ b,
                                                        float* __restrict__ dst, int n) {
  const int i = blockIdx.x * 256 + threadIdx.x;
  if (i < n) dst[i] = a[i] + b[i];
}

// One level of the tree. grid.x = 4^l tiles of 16 rows (rows = tree*n + node).
__global__ __launch_bounds__(32)
void tree_level_kernel(const float* __restrict__ x, const bf16* __restrict__ wb,
                       const float* __restrict__ bsum_l, const float* __restrict__ bsum_r,
                       const float* __restrict__ b_enc, float* __restrict__ out,
                       int lshift, int offs_l, int offs_child, int internal) {
  // All sequence inputs are known up front -> pre-stage everything, one barrier,
  // then the whole 6-step recurrence runs out of LDS + L2-resident weights.
  __shared__ alignas(16) bf16 sHd[16 * 128];   // head
  __shared__ alignas(16) bf16 sC0[16 * 128];   // child 0
  __shared__ alignas(16) bf16 sC1[16 * 128];   // child 1
  __shared__ alignas(16) bf16 sC2[16 * 128];   // child 2
  __shared__ alignas(16) bf16 sC3[16 * 128];   // child 3
  __shared__ alignas(16) bf16 sHa[16 * 128];   // h ping / final hl
  __shared__ alignas(16) bf16 sHb[16 * 128];   // h pong / final hr
  __shared__ alignas(16) bf16 sHc[16 * 128];   // h pong for right LSTM

  const int lane  = threadIdx.x;
  const int rbase = blockIdx.x * 16;

  const bf16* Wih_l = wb;
  const bf16* Whh_l = wb + 65536;
  const bf16* Wih_r = wb + 131072;
  const bf16* Whh_r = wb + 196608;
  const bf16* W_enc = wb + 262144;   // [128][256] bf16

  // ---------------- pre-stage all sequence inputs (loads all in flight together)
  stage_tile(sHd, x, rbase, lshift, offs_l, 0, 0, lane);
  if (internal) {
    stage_tile(sC1, out, rbase, lshift, offs_child, 1, 1, lane);
    stage_tile(sC0, out, rbase, lshift, offs_child, 1, 0, lane);
    stage_tile(sC2, out, rbase, lshift, offs_child, 1, 2, lane);
    stage_tile(sC3, out, rbase, lshift, offs_child, 1, 3, lane);
  }
  __syncthreads();

  v8f cst[8];

  // ---------------- LEFT LSTM: seq = [ch1, ch0, head] (internal) or [head] (leaf)
#pragma unroll
  for (int j = 0; j < 8; ++j) cst[j] = (v8f){};
  if (internal) {
    lstm_step(sC1, nullptr, sHa, Wih_l, Whh_l, bsum_l, cst, lane); __syncthreads();
    lstm_step(sC0, sHa, sHb, Wih_l, Whh_l, bsum_l, cst, lane); __syncthreads();
    lstm_step(sHd, sHb, sHa, Wih_l, Whh_l, bsum_l, cst, lane); __syncthreads();
  } else {
    lstm_step(sHd, nullptr, sHa, Wih_l, Whh_l, bsum_l, cst, lane); __syncthreads();
  }
  // hl is in sHa

  // ---------------- RIGHT LSTM: seq = [head, ch2, ch3] (internal) or [head] (leaf)
#pragma unroll
  for (int j = 0; j < 8; ++j) cst[j] = (v8f){};
  if (internal) {
    lstm_step(sHd, nullptr, sHb, Wih_r, Whh_r, bsum_r, cst, lane); __syncthreads();
    lstm_step(sC2, sHb, sHc, Wih_r, Whh_r, bsum_r, cst, lane); __syncthreads();
    lstm_step(sC3, sHc, sHb, Wih_r, Whh_r, bsum_r, cst, lane); __syncthreads();
  } else {
    lstm_step(sHd, nullptr, sHb, Wih_r, Whh_r, bsum_r, cst, lane); __syncthreads();
  }
  // hr is in sHb

  // ---------------- Encoder: e = sigmoid([hl, hr] @ W_enc^T + b_enc), K = 256
  const int n = lane & 15, g = lane >> 4;
  const int nmask = (1 << lshift) - 1;
#pragma unroll
  for (int jt = 0; jt < 8; ++jt) {
    const float be = b_enc[jt * 16 + n];
    v8f acc = {};
#pragma unroll
    for (int kt = 0; kt < 4; ++kt)
      acc = wmma_bf16(load_a_lds(sHa, kt, lane), load_b_w(W_enc, jt * 16, kt, 256, lane), acc);
#pragma unroll
    for (int kt = 0; kt < 4; ++kt)
      acc = wmma_bf16(load_a_lds(sHb, kt, lane), load_b_w(W_enc, jt * 16, kt + 4, 256, lane), acc);
#pragma unroll
    for (int v = 0; v < 8; ++v) {
      const float ev = sigm(acc[v] + be);
      const int r = rbase + v + 8 * g;   // tile row for C-layout element
      const int t = r >> lshift;
      const int i = r & nmask;
      out[((long)t * NODES + offs_l + i) * 128 + jt * 16 + n] = ev;
    }
  }
}

extern "C" void kernel_launch(void* const* d_in, const int* in_sizes, int n_in,
                              void* d_out, int out_size, void* d_ws, size_t ws_size,
                              hipStream_t stream) {
  const float* x     = (const float*)d_in[0];
  const float* Wih_l = (const float*)d_in[1];
  const float* Whh_l = (const float*)d_in[2];
  const float* bih_l = (const float*)d_in[3];
  const float* bhh_l = (const float*)d_in[4];
  const float* Wih_r = (const float*)d_in[5];
  const float* Whh_r = (const float*)d_in[6];
  const float* bih_r = (const float*)d_in[7];
  const float* bhh_r = (const float*)d_in[8];
  const float* W_enc = (const float*)d_in[9];
  const float* b_enc = (const float*)d_in[10];
  float* out = (float*)d_out;

  // ws layout: 294912 bf16 weights (576 KB), then 1024 f32 fused biases (4 KB)
  bf16*  wb = (bf16*)d_ws;
  float* bs = (float*)((char*)d_ws + 294912 * sizeof(bf16));
  float* bsum_l = bs;
  float* bsum_r = bs + 512;

  // Pre-convert weights to bf16 (L2-resident; read by every wave afterwards)
  convert_w_kernel<<<(65536 + 255) / 256, 256, 0, stream>>>(Wih_l, wb + 0,      65536);
  convert_w_kernel<<<(65536 + 255) / 256, 256, 0, stream>>>(Whh_l, wb + 65536,  65536);
  convert_w_kernel<<<(65536 + 255) / 256, 256, 0, stream>>>(Wih_r, wb + 131072, 65536);
  convert_w_kernel<<<(65536 + 255) / 256, 256, 0, stream>>>(Whh_r, wb + 196608, 65536);
  convert_w_kernel<<<(32768 + 255) / 256, 256, 0, stream>>>(W_enc, wb + 262144, 32768);
  fuse_bias_kernel<<<2, 256, 0, stream>>>(bih_l, bhh_l, bsum_l, 512);
  fuse_bias_kernel<<<2, 256, 0, stream>>>(bih_r, bhh_r, bsum_r, 512);

  // Level offsets offs[l] = (4^l - 1)/3
  int offs[9];
  offs[0] = 0;
  for (int l = 0; l < 8; ++l) offs[l + 1] = offs[l] + (1 << (2 * l));

  // Bottom-up: level 7 (leaves) first; children of level l live in out at offs[l+1]
  for (int l = 7; l >= 0; --l) {
    const int ntiles = 1 << (2 * l);   // (TREES * 4^l) / 16 rows per tile
    const int internal = (l == 7) ? 0 : 1;
    const int offc = (l == 7) ? 0 : offs[l + 1];
    tree_level_kernel<<<ntiles, 32, 0, stream>>>(x, wb, bsum_l, bsum_r, b_enc, out,
                                                 2 * l, offs[l], offc, internal);
  }
}
